// MambaLayer_73624329388581
// MI455X (gfx1250) — matrix operand order repaired and checked by reading, not compiled, not run on hardware
//
#include <hip/hip_runtime.h>
#include <hip/hip_bf16.h>

// ---------------------------------------------------------------------------
// MI455X (gfx1250) windowed tri-directional Mamba block.
//
// Bn=64 windows, L=512 tokens, C=96, d_inner=192, d_state=16, dt_rank=6.
//
//  K0  weight pre-convert f32 -> bf16 (once; B-fragments become pure b128 loads)
//  K1  layernorm                         -> xn    bf16 [tok,96]
//  K2  in_proj  WMMA (1 Mtile x 4 Ntiles/wave) -> xz bf16 [tok,384]
//  K3  causal dw-conv+silu, per branch (time-permuted) -> xconv_br bf16 [tok,192]
//  K4  xproj    WMMA (1x3 tiles/wave), per branch      -> xdbl_br  f32 [tok,48]
//  K5  FUSED 3-branch selective scan: 576 blocks (64 win x 3 br x 3 chunks),
//      dt-projection+softplus inlined (6 MACs), block-uniform scalar loads for
//      dt/B/C rows (no LDS, no barriers), unsafeAtomicAdd merge into out_acc
//  K6  out_proj WMMA (1x6 tiles/wave) + window-reverse + residual -> out5
//  K7-K9 ECA (channel means -> k3 conv -> sigmoid gate -> scale)
//
// Workspace ~126 MB: every intermediate stays resident in the 192 MB L2.
// ---------------------------------------------------------------------------

typedef __attribute__((ext_vector_type(16))) __bf16 v16bf;
typedef __attribute__((ext_vector_type(8)))  float  v8f;

typedef union { v16bf v; unsigned short u[16]; } v16bf_u;

#define BN   64
#define LTOK 512
#define CCH  96
#define DI   192
#define DSTATE 16
#define NTOK (BN * LTOK)   // 32768
#define XDBL_LD 48         // 38 padded to 3 N-tiles

static __device__ __forceinline__ unsigned short f2bf(float f) {
  return __builtin_bit_cast(unsigned short, (__bf16)f);   // hw v_cvt, RNE
}
static __device__ __forceinline__ float bf2f(unsigned short b) {
  unsigned u = ((unsigned)b) << 16;
  return __builtin_bit_cast(float, u);
}
static __device__ __forceinline__ float sigmoidf_(float x) {
  return 1.f / (1.f + expf(-x));
}
// time-permutation: permuted step t -> original token index l
static __device__ __forceinline__ int map_idx(int t, int mode) {
  if (mode == 0) return t;                    // forward
  if (mode == 1) return 511 - t;              // flipped
  return (t & 7) * 64 + (t >> 3);             // inter-slice (NSLICES=8, Ls=64)
}

// ---- WMMA fragment loaders (ISA 7.12.2, bf16 16x16x32) ---------------------
// A: 16x32 (MxK). Per lane: two contiguous 8-elem K runs -> 2x b128 loads.
static __device__ __forceinline__ void
load_a_bf16(v16bf_u& af, const unsigned short* A, int lda, int m0, int k0, int lane) {
  int half = lane >> 4, m = lane & 15;
  const unsigned short* row = A + (size_t)(m0 + m) * lda + k0 + half * 8;
#pragma unroll
  for (int i = 0; i < 8; ++i) { af.u[i] = row[i]; af.u[8 + i] = row[16 + i]; }
}
// A from f32 source (convert on the fly with hw cvt)
static __device__ __forceinline__ void
load_a_f32(v16bf_u& af, const float* A, int lda, int m0, int k0, int lane) {
  int half = lane >> 4, m = lane & 15;
  const float* row = A + (size_t)(m0 + m) * lda + k0 + half * 8;
#pragma unroll
  for (int i = 0; i < 8; ++i) { af.u[i] = f2bf(row[i]); af.u[8 + i] = f2bf(row[16 + i]); }
}
// B: 32x16 (KxN) from bf16 weights stored row-major [N][K].
// Per lane: two contiguous 8-elem K runs (same order as A) -> 2x b128 loads.
static __device__ __forceinline__ void
load_b_bf16(v16bf_u& bf, const unsigned short* W, int ldw, int n0, int k0, int lane) {
  int half = lane >> 4, n = lane & 15;
  const unsigned short* row = W + (size_t)(n0 + n) * ldw + k0 + half * 8;
#pragma unroll
  for (int i = 0; i < 8; ++i) { bf.u[i] = row[i]; bf.u[8 + i] = row[16 + i]; }
}

static __device__ __forceinline__ int tok2vox(int tok) {
  int b = tok >> 9, l = tok & 511;
  int hb = b >> 4, wb = (b >> 2) & 3, db = b & 3;
  int i = l >> 6, j = (l >> 3) & 7, k = l & 7;
  return (hb * 8 + i) * 1024 + (wb * 8 + j) * 32 + (db * 8 + k);
}

// ---- K0: weight f32 -> bf16 (rows >= nvalid zero-padded) -------------------
__global__ void k_cvtw(const float* __restrict__ src, unsigned short* __restrict__ dst,
                       int N, int K, int nvalid) {
  int idx = blockIdx.x * 256 + threadIdx.x;
  if (idx >= N * K) return;
  int n = idx / K, k = idx % K;
  dst[idx] = (n < nvalid) ? f2bf(src[(size_t)n * K + k]) : (unsigned short)0;
}

// ---- K1: layernorm over C=96, one wave per token ---------------------------
__global__ void k_ln(const float* __restrict__ x, const float* __restrict__ lw,
                     const float* __restrict__ lb, unsigned short* __restrict__ xn) {
  int gid  = blockIdx.x * blockDim.x + threadIdx.x;
  int tok  = gid >> 5;
  int lane = gid & 31;
  if (tok >= NTOK) return;
  int vox = tok2vox(tok);
  float v[3];
#pragma unroll
  for (int q = 0; q < 3; ++q) v[q] = x[(size_t)(lane * 3 + q) * 32768 + vox];
  float s = v[0] + v[1] + v[2];
#pragma unroll
  for (int m = 16; m >= 1; m >>= 1) s += __shfl_xor(s, m, 32);
  float mu = s * (1.f / 96.f);
  float s2 = 0.f;
#pragma unroll
  for (int q = 0; q < 3; ++q) { float d = v[q] - mu; s2 += d * d; }
#pragma unroll
  for (int m = 16; m >= 1; m >>= 1) s2 += __shfl_xor(s2, m, 32);
  float rstd = rsqrtf(s2 * (1.f / 96.f) + 1e-5f);
#pragma unroll
  for (int q = 0; q < 3; ++q) {
    int c = lane * 3 + q;
    xn[(size_t)tok * CCH + c] = f2bf((v[q] - mu) * rstd * lw[c] + lb[c]);
  }
}

// ---- K2: in_proj GEMM (M=NTOK, N=384, K=96); 1 Mtile x 4 Ntiles per wave ---
__global__ void k_inproj(const unsigned short* __restrict__ xn,
                         const unsigned short* __restrict__ Wb,
                         unsigned short* __restrict__ xz) {
  int wid  = (blockIdx.x * blockDim.x + threadIdx.x) >> 5;  // 12288 waves
  int lane = threadIdx.x & 31;
  int mt = wid / 6, ng = wid % 6;
  int m0 = mt * 16, n0 = ng * 64;
  v8f acc[4];
#pragma unroll
  for (int nn = 0; nn < 4; ++nn) acc[nn] = (v8f){0.f,0.f,0.f,0.f,0.f,0.f,0.f,0.f};
#pragma unroll
  for (int k0 = 0; k0 < 96; k0 += 32) {
    v16bf_u a;
    load_a_bf16(a, xn, CCH, m0, k0, lane);
#pragma unroll
    for (int nn = 0; nn < 4; ++nn) {
      v16bf_u b;
      load_b_bf16(b, Wb, CCH, n0 + nn * 16, k0, lane);
      acc[nn] = __builtin_amdgcn_wmma_f32_16x16x32_bf16(false, a.v, false, b.v,
                                                        (short)0, acc[nn], false, false);
    }
  }
  int half = lane >> 4, n = lane & 15;
#pragma unroll
  for (int nn = 0; nn < 4; ++nn)
#pragma unroll
    for (int r = 0; r < 8; ++r) {
      int m = m0 + r + half * 8;
      xz[(size_t)m * 384 + n0 + nn * 16 + n] = f2bf(acc[nn][r]);
    }
}

// ---- K3: causal depthwise conv (d_conv=4) + silu, time-permuted ------------
__global__ void k_conv(const unsigned short* __restrict__ xz,
                       const float* __restrict__ cw, const float* __restrict__ cb,
                       unsigned short* __restrict__ xconv, int mode) {
  int idx  = blockIdx.x * 256 + threadIdx.x;       // BN*LTOK*DI threads
  int d    = idx % DI;
  int rest = idx / DI;
  int t    = rest % LTOK;
  int b    = rest / LTOK;
  float acc = cb[d];
#pragma unroll
  for (int k = 0; k < 4; ++k) {
    int tt = t - 3 + k;
    if (tt >= 0) {
      int l = map_idx(tt, mode);
      acc += cw[d * 4 + k] * bf2f(xz[((size_t)(b * LTOK + l)) * 384 + d]);
    }
  }
  float sx = acc * sigmoidf_(acc);
  xconv[((size_t)(b * LTOK + map_idx(t, mode))) * DI + d] = f2bf(sx);
}

// ---- K4: xproj GEMM (M=NTOK, N=38 pad 48, K=192); 1x3 tiles per wave -------
__global__ void k_xdbl(const unsigned short* __restrict__ xconv,
                       const unsigned short* __restrict__ Wb, float* __restrict__ xdbl) {
  int wid  = (blockIdx.x * blockDim.x + threadIdx.x) >> 5;  // 2048 waves
  int lane = threadIdx.x & 31;
  int m0 = wid * 16;
  v8f acc[3];
#pragma unroll
  for (int nn = 0; nn < 3; ++nn) acc[nn] = (v8f){0.f,0.f,0.f,0.f,0.f,0.f,0.f,0.f};
#pragma unroll
  for (int k0 = 0; k0 < DI; k0 += 32) {
    v16bf_u a;
    load_a_bf16(a, xconv, DI, m0, k0, lane);
#pragma unroll
    for (int nn = 0; nn < 3; ++nn) {
      v16bf_u b;
      load_b_bf16(b, Wb, DI, nn * 16, k0, lane);
      acc[nn] = __builtin_amdgcn_wmma_f32_16x16x32_bf16(false, a.v, false, b.v,
                                                        (short)0, acc[nn], false, false);
    }
  }
  int half = lane >> 4, n = lane & 15;
#pragma unroll
  for (int nn = 0; nn < 3; ++nn)
#pragma unroll
    for (int r = 0; r < 8; ++r) {
      int m = m0 + r + half * 8;
      xdbl[(size_t)m * XDBL_LD + nn * 16 + n] = acc[nn][r];
    }
}

// ---- zero-init for the atomic accumulator ----------------------------------
__global__ void k_zero(float* __restrict__ p) {
  p[blockIdx.x * 256 + threadIdx.x] = 0.f;
}

// ---- K5: fused 3-branch selective scan -------------------------------------
struct ScanArgs {
  const unsigned short* xconv[3];
  const float* xdbl[3];
  const float* Wdt[3];
  const float* bdt[3];
  const float* Alog[3];
  const float* Dp[3];
};

__global__ void __launch_bounds__(64)
k_scan(ScanArgs sa, const unsigned short* __restrict__ xz, float* __restrict__ out_acc) {
  int bid   = blockIdx.x;            // 64 win * 3 br * 3 chunks = 576
  int chunk = bid % 3;
  int rest  = bid / 3;
  int b     = rest % BN;
  int br    = rest / BN;
  int d     = chunk * 64 + threadIdx.x;

  const unsigned short* xconv = sa.xconv[br];
  const float* xdbl = sa.xdbl[br];

  float dtw[6], Arow[DSTATE], h[DSTATE];
#pragma unroll
  for (int r = 0; r < 6; ++r) dtw[r] = sa.Wdt[br][d * 6 + r];
  float bias = sa.bdt[br][d];
  float Dd   = sa.Dp[br][d];
#pragma unroll
  for (int s = 0; s < DSTATE; ++s) {
    Arow[s] = -expf(sa.Alog[br][d * DSTATE + s]);
    h[s] = 0.f;
  }

  for (int t = 0; t < LTOK; ++t) {
    int l = map_idx(t, br);
    size_t tokoff = (size_t)(b * LTOK + l);
    const float* row = xdbl + tokoff * XDBL_LD;   // block-uniform -> SMEM loads
    if (t + 1 < LTOK) {
      size_t nx = (size_t)(b * LTOK + map_idx(t + 1, br));
      __builtin_prefetch(&xconv[nx * DI + d], 0, 3);     // global_prefetch_b8
    }
    // inlined dt-projection + softplus
    float dtpre = bias;
#pragma unroll
    for (int r = 0; r < 6; ++r) dtpre += dtw[r] * row[r];
    float dt = (dtpre > 20.f) ? dtpre : logf(1.f + expf(dtpre));

    float u  = bf2f(xconv[tokoff * DI + d]);
    float zv = bf2f(xz[tokoff * 384 + DI + d]);
    float du = dt * u;
    float acc = 0.f;
#pragma unroll
    for (int s = 0; s < DSTATE; ++s) {
      h[s] = h[s] * __expf(dt * Arow[s]) + du * row[6 + s];
      acc += h[s] * row[22 + s];
    }
    float y = (acc + Dd * u) * (zv * sigmoidf_(zv));
    unsafeAtomicAdd(&out_acc[tokoff * DI + d], y);       // global_atomic_add_f32
  }
}

// ---- K6: out_proj GEMM (M=NTOK, N=96, K=192) + window-reverse + residual ---
__global__ void k_outproj(const float* __restrict__ out_acc,
                          const unsigned short* __restrict__ Wb,
                          const float* __restrict__ xin, float* __restrict__ out5) {
  int wid  = (blockIdx.x * blockDim.x + threadIdx.x) >> 5;  // 2048 waves
  int lane = threadIdx.x & 31;
  int m0 = wid * 16;
  v8f acc[6];
#pragma unroll
  for (int nn = 0; nn < 6; ++nn) acc[nn] = (v8f){0.f,0.f,0.f,0.f,0.f,0.f,0.f,0.f};
#pragma unroll
  for (int k0 = 0; k0 < DI; k0 += 32) {
    v16bf_u a;
    load_a_f32(a, out_acc, DI, m0, k0, lane);
#pragma unroll
    for (int nn = 0; nn < 6; ++nn) {
      v16bf_u b;
      load_b_bf16(b, Wb, DI, nn * 16, k0, lane);
      acc[nn] = __builtin_amdgcn_wmma_f32_16x16x32_bf16(false, a.v, false, b.v,
                                                        (short)0, acc[nn], false, false);
    }
  }
  int half = lane >> 4, n = lane & 15;
#pragma unroll
  for (int nn = 0; nn < 6; ++nn) {
    int c = nn * 16 + n;
#pragma unroll
    for (int r = 0; r < 8; ++r) {
      int tok = m0 + r + half * 8;
      int vox = tok2vox(tok);
      size_t gi = (size_t)c * 32768 + vox;
      out5[gi] = acc[nn][r] + xin[gi];
    }
  }
}

// ---- K7-K9: ECA ------------------------------------------------------------
__global__ void k_eca_reduce(const float* __restrict__ out5, float* __restrict__ means) {
  int c = blockIdx.x;
  __shared__ float sm[256];
  float s = 0.f;
  for (int v = threadIdx.x; v < 32768; v += 256) s += out5[(size_t)c * 32768 + v];
  sm[threadIdx.x] = s;
  __syncthreads();
  for (int off = 128; off > 0; off >>= 1) {
    if (threadIdx.x < off) sm[threadIdx.x] += sm[threadIdx.x + off];
    __syncthreads();
  }
  if (threadIdx.x == 0) means[c] = sm[0] * (1.f / 32768.f);
}

__global__ void k_eca_gate(const float* __restrict__ means,
                           const float* __restrict__ w, float* __restrict__ gate) {
  int c = threadIdx.x;
  if (c >= CCH) return;
  float g = w[1] * means[c];
  if (c > 0)       g += w[0] * means[c - 1];
  if (c < CCH - 1) g += w[2] * means[c + 1];
  gate[c] = sigmoidf_(g);
}

__global__ void k_apply(const float* __restrict__ out5, const float* __restrict__ gate,
                        float* __restrict__ out) {
  int idx = blockIdx.x * 256 + threadIdx.x;      // 96*32768 threads
  int c = idx >> 15;
  out[idx] = out5[idx] * gate[c];
}

// ---------------------------------------------------------------------------
extern "C" void kernel_launch(void* const* d_in, const int* in_sizes, int n_in,
                              void* d_out, int out_size, void* d_ws, size_t ws_size,
                              hipStream_t stream) {
  const float* x    = (const float*)d_in[0];
  const float* ln_w = (const float*)d_in[1];
  const float* ln_b = (const float*)d_in[2];
  const float* Wip  = (const float*)d_in[3];   // (384,96)
  const float* Wop  = (const float*)d_in[4];   // (96,192)
  const float* ecaw = (const float*)d_in[5];   // (3,)

  // workspace layout
  size_t off = 0;
  unsigned short* xn = (unsigned short*)((char*)d_ws + off); off += (size_t)NTOK * CCH * 2;
  unsigned short* xz = (unsigned short*)((char*)d_ws + off); off += (size_t)NTOK * 384 * 2;
  unsigned short* xconv[3];
  for (int i = 0; i < 3; ++i) { xconv[i] = (unsigned short*)((char*)d_ws + off); off += (size_t)NTOK * DI * 2; }
  float* xdbl[3];
  for (int i = 0; i < 3; ++i) { xdbl[i] = (float*)((char*)d_ws + off); off += (size_t)NTOK * XDBL_LD * 4; }
  float* oacc = (float*)((char*)d_ws + off); off += (size_t)NTOK * DI * 4;
  float* out5 = (float*)((char*)d_ws + off); off += (size_t)CCH * 32768 * 4;
  unsigned short* WipB = (unsigned short*)((char*)d_ws + off); off += (size_t)384 * CCH * 2;
  unsigned short* WopB = (unsigned short*)((char*)d_ws + off); off += (size_t)CCH * DI * 2;
  unsigned short* WxpB[3];
  for (int i = 0; i < 3; ++i) { WxpB[i] = (unsigned short*)((char*)d_ws + off); off += (size_t)XDBL_LD * DI * 2; }
  float* means = (float*)((char*)d_ws + off); off += 512;
  float* gate  = (float*)((char*)d_ws + off); off += 512;
  if (ws_size < off) return;

  dim3 blk(256);

  // K0: weight conversions
  k_cvtw<<<dim3((384 * 96 + 255) / 256), blk, 0, stream>>>(Wip, WipB, 384, CCH, 384);
  k_cvtw<<<dim3((96 * 192 + 255) / 256), blk, 0, stream>>>(Wop, WopB, CCH, DI, CCH);
  for (int br = 0; br < 3; ++br)
    k_cvtw<<<dim3((XDBL_LD * DI + 255) / 256), blk, 0, stream>>>(
        (const float*)d_in[6 + 7 * br + 2], WxpB[br], XDBL_LD, DI, 38);

  // K1 layernorm
  k_ln<<<dim3(NTOK / 8), blk, 0, stream>>>(x, ln_w, ln_b, xn);

  // K2 in_proj: 2048 Mtiles * 6 Ngroups = 12288 waves
  k_inproj<<<dim3(12288 / 8), blk, 0, stream>>>(xn, WipB, xz);

  // K3/K4 per branch (independent buffers -> back-to-back launches)
  for (int br = 0; br < 3; ++br) {
    k_conv<<<dim3((size_t)NTOK * DI / 256), blk, 0, stream>>>(
        xz, (const float*)d_in[6 + 7 * br + 0], (const float*)d_in[6 + 7 * br + 1],
        xconv[br], br);
    k_xdbl<<<dim3(2048 / 8), blk, 0, stream>>>(xconv[br], WxpB[br], xdbl[br]);
  }

  // K5 fused scan
  k_zero<<<dim3((size_t)NTOK * DI / 256), blk, 0, stream>>>(oacc);
  ScanArgs sa;
  for (int br = 0; br < 3; ++br) {
    sa.xconv[br] = xconv[br];
    sa.xdbl[br]  = xdbl[br];
    sa.Wdt[br]   = (const float*)d_in[6 + 7 * br + 3];
    sa.bdt[br]   = (const float*)d_in[6 + 7 * br + 4];
    sa.Alog[br]  = (const float*)d_in[6 + 7 * br + 5];
    sa.Dp[br]    = (const float*)d_in[6 + 7 * br + 6];
  }
  k_scan<<<dim3(BN * 3 * 3), dim3(64), 0, stream>>>(sa, xz, oacc);

  // K6 out_proj (+ residual, window reverse)
  k_outproj<<<dim3(2048 / 8), blk, 0, stream>>>(oacc, WopB, x, out5);

  // ECA
  k_eca_reduce<<<dim3(CCH), blk, 0, stream>>>(out5, means);
  k_eca_gate<<<dim3(1), dim3(128), 0, stream>>>(means, ecaw, gate);
  k_apply<<<dim3((size_t)CCH * 32768 / 256), blk, 0, stream>>>(out5, gate, (float*)d_out);
}
